// DenseExt_90151363543119
// MI455X (gfx1250) — matrix operand order repaired and checked by reading, not compiled, not run on hardware
//
#include <hip/hip_runtime.h>

// ---------------------------------------------------------------------------
// Dense layer: out[B,N] = A[B,K] @ W[K,N] + bias,  B=16384, K=768, N=768, fp32
// fp32 -> bf16 (hi RNE + lo residual) split, 3x v_wmma_f32_16x16x32_bf16 per
// tile, fp32 accumulation, LDS double-buffered tiles.
// Round 7: 512-thread blocks, 16 waves in a 4x4 grid, 32x32 wave tile
// (acc = 32 VGPRs). Per-thread register footprint roughly halves vs the
// 8-wave version -> comfortably below the 256-VGPR spill cliff.
// ---------------------------------------------------------------------------

typedef __bf16 v16bf __attribute__((ext_vector_type(16)));
typedef float  v8f   __attribute__((ext_vector_type(8)));

#define B_ROWS 16384
#define K_DIM  768
#define N_DIM  768
#define BM 128
#define BN 128
#define BK 32
#define PA 40              // bf16 row pitch in LDS: 32 elements + 8 pad (80B, 16B aligned)
#define NK (K_DIM / BK)    // 24 k-steps
#define NTHREADS 512       // 16 wave32 waves

union Frag { v16bf v; uint4 q[2]; };

static __device__ __forceinline__ void sched_fence_mem() {
#if __has_builtin(__builtin_amdgcn_sched_barrier)
    // allow ALU/VALU/SALU/WMMA to cross, block VMEM & DS motion across this point
    __builtin_amdgcn_sched_barrier(0x0F);
#endif
}

// result = (hiword[31:16] << 16) | loword[31:16]  -- exactly one v_perm_b32
static __device__ __forceinline__ unsigned pack_hi16(unsigned hiword, unsigned loword) {
    unsigned d;
    asm("v_perm_b32 %0, %1, %2, 0x07060302" : "=v"(d) : "v"(hiword), "v"(loword));
    return d;
}

// Split 4 floats into packed bf16 hi (RNE) and packed bf16 lo (truncated residual).
static __device__ __forceinline__ void split4(float x0, float x1, float x2, float x3,
                                              uint2& uh, uint2& ul) {
    unsigned u0 = __builtin_bit_cast(unsigned, x0);
    unsigned u1 = __builtin_bit_cast(unsigned, x1);
    unsigned u2 = __builtin_bit_cast(unsigned, x2);
    unsigned u3 = __builtin_bit_cast(unsigned, x3);
    unsigned r0 = u0 + 0x7FFFu + ((u0 >> 16) & 1u);   // RNE round to bf16 in top 16
    unsigned r1 = u1 + 0x7FFFu + ((u1 >> 16) & 1u);
    unsigned r2 = u2 + 0x7FFFu + ((u2 >> 16) & 1u);
    unsigned r3 = u3 + 0x7FFFu + ((u3 >> 16) & 1u);
    float h0 = __builtin_bit_cast(float, r0 & 0xFFFF0000u);
    float h1 = __builtin_bit_cast(float, r1 & 0xFFFF0000u);
    float h2 = __builtin_bit_cast(float, r2 & 0xFFFF0000u);
    float h3 = __builtin_bit_cast(float, r3 & 0xFFFF0000u);
    unsigned l0 = __builtin_bit_cast(unsigned, x0 - h0);  // residual, |lo| <= 2^-8 |x|
    unsigned l1 = __builtin_bit_cast(unsigned, x1 - h1);
    unsigned l2 = __builtin_bit_cast(unsigned, x2 - h2);
    unsigned l3 = __builtin_bit_cast(unsigned, x3 - h3);
    uh.x = pack_hi16(r1, r0);
    uh.y = pack_hi16(r3, r2);
    ul.x = pack_hi16(l1, l0);   // truncation of residual: err < 2^-16 |x|
    ul.y = pack_hi16(l3, l2);
}

// ---------------------------------------------------------------------------
// Prepass: W[k][n] fp32  ->  Wt_hi[n][k], Wt_lo[n][k] bf16 (transposed).
// ---------------------------------------------------------------------------
__global__ __launch_bounds__(256)
void preconv_w_kernel(const float* __restrict__ W,
                      unsigned short* __restrict__ WtHi,
                      unsigned short* __restrict__ WtLo)
{
    int id = blockIdx.x * 256 + threadIdx.x;       // 0 .. 768*192-1
    int n  = id % N_DIM;
    int kg = id / N_DIM;                           // 0..191 (4 k's each)
    float x0 = W[(size_t)(kg * 4 + 0) * N_DIM + n];
    float x1 = W[(size_t)(kg * 4 + 1) * N_DIM + n];
    float x2 = W[(size_t)(kg * 4 + 2) * N_DIM + n];
    float x3 = W[(size_t)(kg * 4 + 3) * N_DIM + n];
    uint2 uh, ul;
    split4(x0, x1, x2, x3, uh, ul);
    *(uint2*)&WtHi[(size_t)n * K_DIM + kg * 4] = uh;
    *(uint2*)&WtLo[(size_t)n * K_DIM + kg * 4] = ul;
}

// ---------------------------------------------------------------------------
// Main GEMM kernel. PRECONV=true reads bf16 Wt from d_ws (no W conversion in
// the hot loop); PRECONV=false converts W fp32 on the fly (fallback).
// ---------------------------------------------------------------------------
template <bool PRECONV>
__global__ __launch_bounds__(NTHREADS, 1)
void dense_bf16x3_wmma_kernel(const float* __restrict__ A,
                              const float* __restrict__ W,
                              const unsigned short* __restrict__ WtHi,
                              const unsigned short* __restrict__ WtLo,
                              const float* __restrict__ bias,
                              float* __restrict__ out)
{
    // [2 bufs] x {A_hi, A_lo} (row-major [m][k]) and {Wt_hi, Wt_lo} ([n][k])
    __shared__ __align__(16) unsigned short smem[8 * BM * PA];   // 80 KB
    unsigned short* sAhi = smem + 0 * BM * PA;
    unsigned short* sAlo = smem + 2 * BM * PA;
    unsigned short* sWhi = smem + 4 * BM * PA;
    unsigned short* sWlo = smem + 6 * BM * PA;

    const int t    = threadIdx.x;
    const int lane = t & 31;
    const int wave = t >> 5;        // 0..15
    const int wm   = wave & 3;      // 4 waves along M, 32 rows each
    const int wn   = wave >> 2;     // 4 waves along N, 32 cols each
    const int r16  = lane & 15;
    const int half = lane >> 4;

    const int rowBase = blockIdx.y * BM;
    const int colBase = blockIdx.x * BN;

    // ---- thread-invariant 32-bit offsets (computed once) ----
    unsigned aGOff[2], aSOff[2];          // A: global (elements), LDS (shorts)
    #pragma unroll
    for (int i = 0; i < 2; ++i) {
        int f = t + i * NTHREADS; int row = f >> 3; int c4 = f & 7;
        aGOff[i] = (unsigned)(row * K_DIM + c4 * 4);
        aSOff[i] = (unsigned)(row * PA + c4 * 4);
    }
    // W preconv: one uint4 per thread per (hi|lo) tile: 512 uint4 each
    const int wRow = t >> 2;              // 0..127
    const int wC   = t & 3;               // 4 uint4 per row
    const unsigned wGOff = (unsigned)((colBase + wRow) * K_DIM + wC * 8);
    const unsigned wSOff = (unsigned)(wRow * PA + wC * 8);
    // W fallback mapping
    const int wn_col = t & 127;
    const int wkg0   = t >> 7;            // 0..3

    // A staging registers for the in-flight tile (held across compute)
    float4 av[2];

    auto load_A = [&](int ks) {
        const float* Abase = A + (size_t)rowBase * K_DIM + ks * BK;   // uniform
        #pragma unroll
        for (int i = 0; i < 2; ++i)
            av[i] = *(const float4*)(Abase + aGOff[i]);
    };

    // Stage one k-step: issue W loads first (L2-resident), overlap with the
    // A fp32->bf16 split, then store W to LDS.  All registers short-lived.
    auto stage_tile = [&](int ks, int buf) {
        const unsigned bofs = (unsigned)buf * (BM * PA);
        unsigned short* dAh = sAhi + bofs;
        unsigned short* dAl = sAlo + bofs;
        unsigned short* dWh = sWhi + bofs;
        unsigned short* dWl = sWlo + bofs;
        if constexpr (PRECONV) {
            uint4 wqh = *(const uint4*)(WtHi + ks * BK + wGOff);
            uint4 wql = *(const uint4*)(WtLo + ks * BK + wGOff);
            #pragma unroll
            for (int i = 0; i < 2; ++i) {                  // A split hides W latency
                uint2 uh, ul;
                split4(av[i].x, av[i].y, av[i].z, av[i].w, uh, ul);
                *(uint2*)&dAh[aSOff[i]] = uh;
                *(uint2*)&dAl[aSOff[i]] = ul;
            }
            *(uint4*)(dWh + wSOff) = wqh;
            *(uint4*)(dWl + wSOff) = wql;
        } else {
            float wv[2][4];
            const float* Ws = W + (size_t)ks * BK * N_DIM + colBase;  // uniform
            #pragma unroll
            for (int i = 0; i < 2; ++i) {
                int kg = wkg0 + i * 4;
                #pragma unroll
                for (int j = 0; j < 4; ++j)
                    wv[i][j] = Ws[(unsigned)((kg * 4 + j) * N_DIM + wn_col)];
            }
            #pragma unroll
            for (int i = 0; i < 2; ++i) {
                uint2 uh, ul;
                split4(av[i].x, av[i].y, av[i].z, av[i].w, uh, ul);
                *(uint2*)&dAh[aSOff[i]] = uh;
                *(uint2*)&dAl[aSOff[i]] = ul;
            }
            #pragma unroll
            for (int i = 0; i < 2; ++i) {
                int kg = wkg0 + i * 4;
                uint2 uh, ul;
                split4(wv[i][0], wv[i][1], wv[i][2], wv[i][3], uh, ul);
                *(uint2*)&dWh[(unsigned)(wn_col * PA + kg * 4)] = uh;
                *(uint2*)&dWl[(unsigned)(wn_col * PA + kg * 4)] = ul;
            }
        }
    };

    v8f acc[2][2];
    {
        v8f z = {0.f, 0.f, 0.f, 0.f, 0.f, 0.f, 0.f, 0.f};
        #pragma unroll
        for (int mt = 0; mt < 2; ++mt)
            #pragma unroll
            for (int nt = 0; nt < 2; ++nt) acc[mt][nt] = z;
    }

    load_A(0);
    stage_tile(0, 0);

    int p = 0;
    for (int ks = 0; ks < NK; ++ks) {
        __syncthreads();
        const bool more = (ks + 1 < NK);
        if (more) load_A(ks + 1);            // HBM stream: issue early, hide behind WMMA

        const unsigned pofs = (unsigned)p * (BM * PA);
        const unsigned short* pAh = sAhi + pofs;
        const unsigned short* pAl = sAlo + pofs;
        const unsigned short* pWh = sWhi + pofs;
        const unsigned short* pWl = sWlo + pofs;

        Frag ah[2], al[2];
        #pragma unroll
        for (int mt = 0; mt < 2; ++mt) {
            // A 16x32 bf16 layout: lane r16 = row; K runs [kb..kb+7],[kb+16..kb+23], kb = half*8
            const unsigned short* ba = pAh + (wm * 32 + mt * 16 + r16) * PA + half * 8;
            const unsigned short* bl = pAl + (wm * 32 + mt * 16 + r16) * PA + half * 8;
            ah[mt].q[0] = *(const uint4*)(ba);
            ah[mt].q[1] = *(const uint4*)(ba + 16);
            al[mt].q[0] = *(const uint4*)(bl);
            al[mt].q[1] = *(const uint4*)(bl + 16);
        }
        // ---- pass 1: B-hi products (2 WMMAs per fragment; al dies after this) ----
        #pragma unroll
        for (int nt = 0; nt < 2; ++nt) {
            const unsigned short* bb = pWh + (wn * 32 + nt * 16 + r16) * PA + half * 8;
            Frag bh;
            bh.q[0] = *(const uint4*)(bb);
            bh.q[1] = *(const uint4*)(bb + 16);
            #pragma unroll
            for (int mt = 0; mt < 2; ++mt) {
                acc[mt][nt] = __builtin_amdgcn_wmma_f32_16x16x32_bf16(
                    false, ah[mt].v, false, bh.v, (short)0, acc[mt][nt], false, false);
                acc[mt][nt] = __builtin_amdgcn_wmma_f32_16x16x32_bf16(
                    false, al[mt].v, false, bh.v, (short)0, acc[mt][nt], false, false);
            }
        }
        // ---- pass 2: B-lo products (1 WMMA per mt per fragment) ----
        #pragma unroll
        for (int nt = 0; nt < 2; ++nt) {
            const unsigned short* bv = pWl + (wn * 32 + nt * 16 + r16) * PA + half * 8;
            Frag blo;
            blo.q[0] = *(const uint4*)(bv);
            blo.q[1] = *(const uint4*)(bv + 16);
            #pragma unroll
            for (int mt = 0; mt < 2; ++mt) {
                acc[mt][nt] = __builtin_amdgcn_wmma_f32_16x16x32_bf16(
                    false, ah[mt].v, false, blo.v, (short)0, acc[mt][nt], false, false);
            }
        }

        if (more) {
            sched_fence_mem();               // keep staging VMEM/DS out of compute region
            stage_tile(ks + 1, 1 - p);
            p ^= 1;
        }
    }

    // ---- epilogue: C/D layout VGPR r -> row r (lanes 0-15) / r+8 (lanes 16-31) ----
    #pragma unroll
    for (int mt = 0; mt < 2; ++mt) {
        #pragma unroll
        for (int nt = 0; nt < 2; ++nt) {
            const int gm = rowBase + wm * 32 + mt * 16 + half * 8;
            const int gn = colBase + wn * 32 + nt * 16 + r16;
            const float bvv = bias[gn];
            float* op = out + (size_t)gm * N_DIM + gn;
            #pragma unroll
            for (int rr = 0; rr < 8; ++rr)
                op[(size_t)rr * N_DIM] = acc[mt][nt][rr] + bvv;
        }
    }
}

extern "C" void kernel_launch(void* const* d_in, const int* in_sizes, int n_in,
                              void* d_out, int out_size, void* d_ws, size_t ws_size,
                              hipStream_t stream) {
    const float* A    = (const float*)d_in[0];   // inputs [16384,768]
    const float* W    = (const float*)d_in[1];   // kernel [768,768]
    const float* bias = (const float*)d_in[2];   // bias   [768]
    float* out = (float*)d_out;                  // [16384,768]

    dim3 grid(N_DIM / BN, B_ROWS / BM);          // (6, 128)
    dim3 block(NTHREADS);                        // 16 wave32 waves

    const size_t WS_NEEDED = (size_t)2 * K_DIM * N_DIM * sizeof(unsigned short); // 2.36 MB
    if (d_ws != nullptr && ws_size >= WS_NEEDED) {
        unsigned short* WtHi = (unsigned short*)d_ws;
        unsigned short* WtLo = WtHi + (size_t)K_DIM * N_DIM;
        const int preBlocks = (N_DIM * (K_DIM / 4)) / 256;   // 576
        hipLaunchKernelGGL(preconv_w_kernel, dim3(preBlocks), dim3(256), 0, stream,
                           W, WtHi, WtLo);
        hipLaunchKernelGGL((dense_bf16x3_wmma_kernel<true>), grid, block, 0, stream,
                           A, W, WtHi, WtLo, bias, out);
    } else {
        hipLaunchKernelGGL((dense_bf16x3_wmma_kernel<false>), grid, block, 0, stream,
                           A, W, nullptr, nullptr, bias, out);
    }
}
